// Generator_67388036874699
// MI455X (gfx1250) — matrix operand order repaired and checked by reading, not compiled, not run on hardware
//
#include <hip/hip_runtime.h>

// LSTM: B=4096, E=H=512, 4H=2048, SEQ=128.
#define B_   4096
#define E_   512
#define H_   512
#define G_   2048
#define SEQ_ 128

typedef __attribute__((ext_vector_type(16))) _Float16 v16h;
typedef __attribute__((ext_vector_type(8)))  _Float16 v8h;
typedef __attribute__((ext_vector_type(8)))  float    v8f;
typedef __attribute__((vector_size(16)))     int      v4i_t;

union V16H { v16h v; v8h h[2]; };

__device__ __forceinline__ v8f wmma_f16(v16h a, v16h b, v8f c) {
  // D(f32,16x16) = A(f16,16x32) * B(f16,32x16) + C(f32)
  return __builtin_amdgcn_wmma_f32_16x16x32_f16(false, a, false, b, (short)0, c,
                                                false, false);
}

__device__ __forceinline__ float fexp2(float x) {
#if __has_builtin(__builtin_amdgcn_exp2f)
  return __builtin_amdgcn_exp2f(x);
#else
  return exp2f(x);
#endif
}
__device__ __forceinline__ float fsigmoid(float x) {
  return 1.0f / (1.0f + fexp2(-1.44269504f * x));
}
__device__ __forceinline__ float ftanh(float x) {
#if __has_builtin(__builtin_amdgcn_tanhf)
  return __builtin_amdgcn_tanhf(x);
#else
  float e = fexp2(2.88539008f * x);   // exp(2x)
  return (e - 1.0f) / (e + 1.0f);
#endif
}

// ---------------- Phase 0: f32 -> f16 conversions -------------------------
__global__ void lstm_convert(const float* __restrict__ x,
                             const float* __restrict__ wih,
                             const float* __restrict__ whh,
                             _Float16* __restrict__ x16,
                             _Float16* __restrict__ wih16,
                             _Float16* __restrict__ whh16) {
  int i = blockIdx.x * blockDim.x + threadIdx.x;
  if (i < B_ * E_) x16[i] = (_Float16)x[i];
  if (i < G_ * H_) {
    wih16[i] = (_Float16)wih[i];
    whh16[i] = (_Float16)whh[i];
  }
}

// ---------------- Phase 1: x_proj = x @ w_ih^T + b_ih + b_hh --------------
// One 16x16 tile per wave. Result stored f16 in FRAGMENT-MAJOR layout:
//   xproj16[((mtile*128 + ntile)*32 + lane)*8 + r]  (one b128 per lane)
// so phase 2 can bulk async-copy it and re-load each C tile as one b128.
__global__ void lstm_xproj(const _Float16* __restrict__ x16,
                           const _Float16* __restrict__ wih16,
                           const float* __restrict__ bih,
                           const float* __restrict__ bhh,
                           _Float16* __restrict__ xproj16) {
  const int tid  = threadIdx.x;
  const int lane = tid & 31, wave = tid >> 5;
  const int col  = lane & 15, kr = lane >> 4;
  const int mt   = blockIdx.x;                 // 0..255
  const int nt   = blockIdx.y * 8 + wave;      // 0..127
  const int m0   = mt * 16;
  const int n0   = nt * 16;

  const float bsum = bih[n0 + col] + bhh[n0 + col];
  v8f acc;
#pragma unroll
  for (int r = 0; r < 8; ++r) acc[r] = bsum;

  const _Float16* arow = x16 + (size_t)(m0 + col) * E_ + kr * 16;
  const _Float16* brow = wih16 + (size_t)(n0 + col) * E_ + kr * 16;
#pragma unroll 4
  for (int k0 = 0; k0 < E_; k0 += 32) {
    V16H a;
    a.h[0] = *(const v8h*)(arow + k0);        // K = k0 + kr*16 .. +7
    a.h[1] = *(const v8h*)(arow + k0 + 8);    // K = k0+kr*16+8 .. +15
    v16h b = *(const v16h*)(brow + k0);       // 16 contiguous K at column n0+col
    acc = wmma_f16(a.v, b, acc);
  }
  v8h d;
#pragma unroll
  for (int r = 0; r < 8; ++r) d[r] = (_Float16)acc[r];
  *(v8h*)(xproj16 + (((size_t)mt * 128 + nt) * 32 + lane) * 8) = d;
}

// ---------------- Phase 2: 128-step recurrence ----------------------------
// 32 batch rows per workgroup (halves per-step L2 w_hh traffic vs 16), 8
// waves. Wave w owns h-columns [64w,64w+64): 4 column tiles x 2 row tiles,
// with matching i/f/g/o gate tiles, so the cell update is lane-local.
// h is double-buffered in LDS (one barrier/step); c stays in registers;
// x_proj fragments are async-copied to LDS once; w_hh streams from L2 and
// each B fragment is reused by 2 WMMAs (both row tiles).
__global__ void lstm_recurrence(const _Float16* __restrict__ xproj16,
                                const float* __restrict__ h0,
                                const float* __restrict__ c0,
                                const _Float16* __restrict__ whh16,
                                float* __restrict__ out,
                                int seq_len) {
  extern __shared__ char smem[];
  _Float16* lds_xp = (_Float16*)smem;                    // 2*128*32*8 halves (128KB)
  _Float16* lds_h  = (_Float16*)(smem + 128 * 1024);     // 2 x (32*512) halves (64KB)

  const int tid  = threadIdx.x;
  const int lane = tid & 31, wave = tid >> 5;
  const int col  = lane & 15, kr = lane >> 4;
  const int mb   = blockIdx.x;          // 0..127
  const int m0   = mb * 32;
  const int nh0  = wave * 64;

  // ---- async-stage this WG's x_proj fragments (contiguous 128KB) ----
  const _Float16* xp_src = xproj16 + (size_t)(mb * 2) * 128 * 32 * 8;
#if __has_builtin(__builtin_amdgcn_global_load_async_to_lds_b128) && \
    __has_builtin(__builtin_amdgcn_s_wait_asynccnt)
  {
    typedef __attribute__((address_space(1))) v4i_t GV4;
    typedef __attribute__((address_space(3))) v4i_t LV4;
    GV4* gsrc = (GV4*)xp_src;
    LV4* ldst = (LV4*)lds_xp;
    for (int i = tid; i < 8192; i += 256)   // 8192 x b128 = 128KB
      __builtin_amdgcn_global_load_async_to_lds_b128(gsrc + i, ldst + i, 0, 0);
    __builtin_amdgcn_s_wait_asynccnt(0);
  }
#else
  {
    const float4* s4 = (const float4*)xp_src;
    float4* d4 = (float4*)lds_xp;
    for (int i = tid; i < 8192; i += 256) d4[i] = s4[i];
  }
#endif

  // h0 -> f16 into buffer 0
  for (int i = tid; i < 32 * H_; i += 256)
    lds_h[i] = (_Float16)h0[(size_t)m0 * H_ + i];

  // c0 -> registers in C/D fragment layout  [rowtile][t]
  v8f cfrag[2][4];
#pragma unroll
  for (int rt = 0; rt < 2; ++rt)
#pragma unroll
    for (int t = 0; t < 4; ++t)
#pragma unroll
      for (int r = 0; r < 8; ++r)
        cfrag[rt][t][r] =
            c0[(size_t)(m0 + rt * 16 + kr * 8 + r) * H_ + nh0 + t * 16 + col];

  __syncthreads();

  for (int step = 0; step < seq_len; ++step) {
    const _Float16* hread = lds_h + (step & 1) * (32 * H_);
    _Float16* hwrite      = lds_h + ((step + 1) & 1) * (32 * H_);
    const bool last = (step == seq_len - 1);

#pragma unroll
    for (int t = 0; t < 4; ++t) {
      const int ntb = wave * 4 + t;     // column tile index within a gate group
      // C-init: one ds_load_b128 per (rowtile, gate) tile
      v8f acc[2][4];
#pragma unroll
      for (int rt = 0; rt < 2; ++rt)
#pragma unroll
        for (int g = 0; g < 4; ++g) {
          v8h cc = *(const v8h*)(lds_xp + ((rt * 128 + g * 32 + ntb) * 32 + lane) * 8);
#pragma unroll
          for (int r = 0; r < 8; ++r) acc[rt][g][r] = (float)cc[r];
        }

      const _Float16* bbase = whh16 + (size_t)(nh0 + t * 16 + col) * H_ + kr * 16;
      const _Float16* hr0 = hread + col * H_ + kr * 16;         // rowtile 0, M=col
      const _Float16* hr1 = hread + (16 + col) * H_ + kr * 16;  // rowtile 1
#pragma unroll 2
      for (int k0 = 0; k0 < H_; k0 += 32) {
        V16H a0, a1;
        a0.h[0] = *(const v8h*)(hr0 + k0);
        a0.h[1] = *(const v8h*)(hr0 + k0 + 8);
        a1.h[0] = *(const v8h*)(hr1 + k0);
        a1.h[1] = *(const v8h*)(hr1 + k0 + 8);
        v16h b0 = *(const v16h*)(bbase + (size_t)0 * 512 * H_ + k0);
        v16h b1 = *(const v16h*)(bbase + (size_t)1 * 512 * H_ + k0);
        v16h b2 = *(const v16h*)(bbase + (size_t)2 * 512 * H_ + k0);
        v16h b3 = *(const v16h*)(bbase + (size_t)3 * 512 * H_ + k0);
        acc[0][0] = wmma_f16(a0.v, b0, acc[0][0]);   // each B reused by both
        acc[1][0] = wmma_f16(a1.v, b0, acc[1][0]);   // row tiles
        acc[0][1] = wmma_f16(a0.v, b1, acc[0][1]);
        acc[1][1] = wmma_f16(a1.v, b1, acc[1][1]);
        acc[0][2] = wmma_f16(a0.v, b2, acc[0][2]);
        acc[1][2] = wmma_f16(a1.v, b2, acc[1][2]);
        acc[0][3] = wmma_f16(a0.v, b3, acc[0][3]);
        acc[1][3] = wmma_f16(a1.v, b3, acc[1][3]);
      }

      // lane-local LSTM cell update; new h goes straight to the write buffer
#pragma unroll
      for (int rt = 0; rt < 2; ++rt)
#pragma unroll
        for (int r = 0; r < 8; ++r) {
          float iv = fsigmoid(acc[rt][0][r]);
          float fv = fsigmoid(acc[rt][1][r]);
          float gv = ftanh(acc[rt][2][r]);
          float ov = fsigmoid(acc[rt][3][r]);
          float cv = fv * cfrag[rt][t][r] + iv * gv;
          cfrag[rt][t][r] = cv;
          float hv = ov * ftanh(cv);
          hwrite[(rt * 16 + kr * 8 + r) * H_ + nh0 + t * 16 + col] = (_Float16)hv;
          if (last) {
            size_t off = (size_t)(m0 + rt * 16 + kr * 8 + r) * H_ +
                         nh0 + t * 16 + col;
            out[off] = hv;                       // h_final (f32)
            out[(size_t)B_ * H_ + off] = cv;     // c_final (f32)
          }
        }
    }
    __syncthreads();   // reads of hread done; hwrite visible to all waves
  }
}

// ---------------------------------------------------------------------------
extern "C" void kernel_launch(void* const* d_in, const int* in_sizes, int n_in,
                              void* d_out, int out_size, void* d_ws, size_t ws_size,
                              hipStream_t stream) {
  (void)in_sizes; (void)n_in; (void)out_size; (void)ws_size;
  const float* start_emb = (const float*)d_in[0];   // (B,1,E)
  const float* h0        = (const float*)d_in[1];   // (1,B,H)
  const float* c0        = (const float*)d_in[2];   // (1,B,H)
  const float* w_ih      = (const float*)d_in[3];   // (4H,E)
  const float* w_hh      = (const float*)d_in[4];   // (4H,H)
  const float* b_ih      = (const float*)d_in[5];   // (4H,)
  const float* b_hh      = (const float*)d_in[6];   // (4H,)

  // workspace: [x_proj f16 frag-major 16MB | w_hh f16 2MB | w_ih f16 2MB | x f16 4MB]
  _Float16* xproj16 = (_Float16*)d_ws;
  _Float16* whh16   = xproj16 + (size_t)B_ * G_;
  _Float16* wih16   = whh16 + (size_t)G_ * H_;
  _Float16* x16     = wih16 + (size_t)G_ * E_;
  float* out = (float*)d_out;

  lstm_convert<<<dim3((B_ * E_ + 255) / 256), dim3(256), 0, stream>>>(
      start_emb, w_ih, w_hh, x16, wih16, whh16);

  lstm_xproj<<<dim3(B_ / 16, G_ / 16 / 8), dim3(256), 0, stream>>>(
      x16, wih16, b_ih, b_hh, xproj16);

  size_t lds_bytes = 128 * 1024 + 2 * 32 * H_ * sizeof(_Float16);  // 192 KB
  (void)hipFuncSetAttribute((const void*)lstm_recurrence,
                            hipFuncAttributeMaxDynamicSharedMemorySize,
                            (int)lds_bytes);
  lstm_recurrence<<<dim3(B_ / 32), dim3(256), lds_bytes, stream>>>(
      xproj16, h0, c0, whh16, out, SEQ_);
}
